// Hungarian_Node_align_Node_loss_34505767256125
// MI455X (gfx1250) — compile-verified, matrix operand order checked by reading, not run on hardware
//
#include <hip/hip_runtime.h>

// ---------------------------------------------------------------- types
typedef __bf16 bf16;
typedef __attribute__((ext_vector_type(8)))  bf16  v8bf;
typedef __attribute__((ext_vector_type(16))) bf16  v16bf;
typedef __attribute__((ext_vector_type(8)))  float v8f;

// ---------------------------------------------------------------- sizes
#define BB     48
#define QSZ    40
#define CSZ    48
#define MAXQ   48
#define DEG    12
#define DNIN   32
#define DEIN   16
#define DD     256
#define DEH    64
#define MSGD   384
#define NPROP  3
#define NNODE  (BB*(QSZ+CSZ))   // 4224
#define EDG    (DEG*NNODE)      // 50688
#define NIND   (2*BB*QSZ)       // 3840
#define ECHUNK 8448
#define NCHUNK 6
#define INVTEMP 10.0f
#define SKIT   20

// ======================================================== WMMA fragment load
// 16-bit A/B fragment: lane L -> row (L&15), half (L>>4).
// element j(0..15): K = k0 + (j>=8?16:0) + half*8 + (j&7)  -> two 16B chunks.
__device__ __forceinline__ v16bf frag_ld(const bf16* base, long ld, int row,
                                         int k0, int half) {
  const bf16* p = base + (long)row * ld + k0 + half * 8;
  union { v16bf v; v8bf h[2]; } u;
  u.h[0] = *(const v8bf*)p;
  u.h[1] = *(const v8bf*)(p + 16);
  return u.v;
}

// ======================================================== generic WMMA GEMM
// Y(M,N) = epilogue( A(M,K)bf16 @ Bt(N,K)^T bf16 + bias )
// mode 0: store f32   1: store bf16 (relu optional)
//      2: atomicAdd into ((float*)Y)[sidx[row]*N + col] * (smask?smask[row]:1)
//      3: resid[row*N+col] += v   (residual update, Y unused)
#define GWAVES 4
__global__ void hk_gemm(const bf16* __restrict__ A, const bf16* __restrict__ Bt,
                        const float* __restrict__ bias, int M, int N, int K,
                        void* Yv, int mode, int relu,
                        const int* __restrict__ sidx,
                        const float* __restrict__ smask,
                        float* __restrict__ resid) {
  int wave = threadIdx.x >> 5, lane = threadIdx.x & 31;
  int m0 = blockIdx.x * 16;
  int n0 = (blockIdx.y * GWAVES + wave) * 16;
  if (n0 >= N) return;
  int half = lane >> 4, q = lane & 15;
  v8f acc = {0.f, 0.f, 0.f, 0.f, 0.f, 0.f, 0.f, 0.f};
  const bf16* arow = A + (long)(m0 + q) * K;
  for (int k0 = 0; k0 < K; k0 += 32) {
    if (k0 + 32 < K) __builtin_prefetch(arow + k0 + 32 + half * 8);
    v16bf af = frag_ld(A, K, m0 + q, k0, half);
    v16bf bf = frag_ld(Bt, K, n0 + q, k0, half);
    acc = __builtin_amdgcn_wmma_f32_16x16x32_bf16(false, af, false, bf,
                                                  (short)0, acc, false, false);
  }
  int col = n0 + q;
  int rb = m0 + half * 8;
  float bv = bias ? bias[col] : 0.f;
  for (int r = 0; r < 8; ++r) {
    int row = rb + r;
    float v = acc[r] + bv;
    if (relu) v = fmaxf(v, 0.f);
    if (mode == 0) {
      ((float*)Yv)[(long)row * N + col] = v;
    } else if (mode == 1) {
      ((bf16*)Yv)[(long)row * N + col] = (bf16)v;
    } else if (mode == 2) {
      float s = smask ? smask[row] : 1.f;
      atomicAdd(&((float*)Yv)[(long)sidx[row] * N + col], v * s);
    } else {
      resid[(long)row * N + col] += v;
    }
  }
}

// batched NT GEMM: out[b](M,N)f32 = A[b](M,K) @ Bm[b](N,K)^T   (bf16 in)
__global__ void hk_gemm_nt(const bf16* __restrict__ A, const bf16* __restrict__ Bm,
                           float* __restrict__ out, int M, int N, int K,
                           long sA, long sB, long sO) {
  int wave = threadIdx.x >> 5, lane = threadIdx.x & 31;
  int m0 = blockIdx.x * 16;
  int n0 = (blockIdx.y * GWAVES + wave) * 16;
  if (n0 >= N) return;
  const bf16* Ab = A + (long)blockIdx.z * sA;
  const bf16* Bb = Bm + (long)blockIdx.z * sB;
  float* Ob = out + (long)blockIdx.z * sO;
  int half = lane >> 4, q = lane & 15;
  v8f acc = {0.f, 0.f, 0.f, 0.f, 0.f, 0.f, 0.f, 0.f};
  for (int k0 = 0; k0 < K; k0 += 32) {
    v16bf af = frag_ld(Ab, K, m0 + q, k0, half);
    v16bf bf = frag_ld(Bb, K, n0 + q, k0, half);
    acc = __builtin_amdgcn_wmma_f32_16x16x32_bf16(false, af, false, bf,
                                                  (short)0, acc, false, false);
  }
  int col = n0 + q, rb = m0 + half * 8;
  for (int r = 0; r < 8; ++r) Ob[(long)(rb + r) * N + col] = acc[r];
}

// ======================================================== small kernels
__global__ void hk_castWT(const float* __restrict__ W, bf16* __restrict__ Wt,
                          int K, int N, int Kpad, int total) {
  int i = blockIdx.x * blockDim.x + threadIdx.x;
  if (i >= total) return;
  int n = i / Kpad, k = i % Kpad;
  Wt[i] = (k < K) ? (bf16)W[(long)k * N + n] : (bf16)0.f;
}

__global__ void hk_castpad(const float* __restrict__ X, bf16* __restrict__ Y,
                           int K, int Kpad, int total) {
  int i = blockIdx.x * blockDim.x + threadIdx.x;
  if (i >= total) return;
  int m = i / Kpad, k = i % Kpad;
  Y[i] = (k < K) ? (bf16)X[(long)m * K + k] : (bf16)0.f;
}

// concat [h[a](256) , h[b](256) , e(64)] -> bf16 (ecnt x 576)
__global__ void hk_concat_e(const float* __restrict__ h, const bf16* __restrict__ e,
                            const int* __restrict__ fi, const int* __restrict__ ti,
                            bf16* __restrict__ C, int e0, int swap, int total) {
  int i = blockIdx.x * blockDim.x + threadIdx.x;
  if (i >= total) return;
  int ed = i / 576, c = i % 576;
  int ge = e0 + ed;
  int a = fi[ge], b = ti[ge];
  if (swap) { int t = a; a = b; b = t; }
  bf16 v;
  if (c < 256)      v = (bf16)h[(long)a * 256 + c];
  else if (c < 512) v = (bf16)h[(long)b * 256 + (c - 256)];
  else              v = e[(long)ge * 64 + (c - 512)];
  C[i] = v;
}

// concat [h(256), agg(384)] -> bf16 (Nn x 640)
__global__ void hk_concat_n(const float* __restrict__ h, const float* __restrict__ agg,
                            bf16* __restrict__ U, int total) {
  int i = blockIdx.x * blockDim.x + threadIdx.x;
  if (i >= total) return;
  int r = i / 640, c = i % 640;
  float v = (c < 256) ? h[(long)r * 256 + c] : agg[(long)r * 384 + (c - 256)];
  U[i] = (bf16)v;
}

// build padded qs/cs (B*48,256) f32 + bf16 from h(B*rpb,256)
__global__ void hk_buildqc(const float* __restrict__ h, float* __restrict__ of,
                           bf16* __restrict__ ob, int rpb, int roff, int valid,
                           int total) {
  int i = blockIdx.x * blockDim.x + threadIdx.x;
  if (i >= total) return;
  int row = i / 256, c = i % 256;
  int b = row / 48, r = row % 48;
  float v = (r < valid) ? h[((long)b * rpb + roff + r) * 256 + c] : 0.f;
  of[i] = v;
  ob[i] = (bf16)v;
}

__global__ void hk_maskcast(const float* __restrict__ X, bf16* __restrict__ Y,
                            int vrows, int total) {
  int i = blockIdx.x * blockDim.x + threadIdx.x;
  if (i >= total) return;
  int r = (i / 256) % 48;
  Y[i] = (r < vrows) ? (bf16)X[i] : (bf16)0.f;
}

// plan f32 (B,48,48) -> bf16 (B,48,64) zero padded K
__global__ void hk_planpad(const float* __restrict__ P, bf16* __restrict__ Pb,
                           int total) {
  int i = blockIdx.x * blockDim.x + threadIdx.x;
  if (i >= total) return;
  int b = i / 3072, rem = i % 3072, r = rem / 64, k = rem % 64;
  Pb[i] = (k < 48) ? (bf16)P[(long)b * 2304 + r * 48 + k] : (bf16)0.f;
}

// cs^T bf16 (B,256,64) from cs_f (B*48,256)
__global__ void hk_cst(const float* __restrict__ cs, bf16* __restrict__ T, int total) {
  int i = blockIdx.x * blockDim.x + threadIdx.x;
  if (i >= total) return;
  int b = i / 16384, rem = i % 16384, d = rem / 64, c = rem % 64;
  T[i] = (c < 48) ? (bf16)cs[((long)b * 48 + c) * 256 + d] : (bf16)0.f;
}

__global__ void hk_sinkhorn(const float* __restrict__ S, float* __restrict__ P) {
  __shared__ float la[48 * 48];
  int b = blockIdx.x, t = threadIdx.x;
  for (int i = t; i < 2304; i += 64) la[i] = S[(long)b * 2304 + i] * INVTEMP;
  __syncthreads();
  for (int it = 0; it < SKIT; ++it) {
    if (t < 48) {                       // axis=2: over cols within a row
      float m = -3.0e38f;
      for (int j = 0; j < 48; ++j) m = fmaxf(m, la[t * 48 + j]);
      float s = 0.f;
      for (int j = 0; j < 48; ++j) s += expf(la[t * 48 + j] - m);
      float l = m + logf(s);
      for (int j = 0; j < 48; ++j) la[t * 48 + j] -= l;
    }
    __syncthreads();
    if (t < 48) {                       // axis=1: over rows within a col
      float m = -3.0e38f;
      for (int i = 0; i < 48; ++i) m = fmaxf(m, la[i * 48 + t]);
      float s = 0.f;
      for (int i = 0; i < 48; ++i) s += expf(la[i * 48 + t] - m);
      float l = m + logf(s);
      for (int i = 0; i < 48; ++i) la[i * 48 + t] -= l;
    }
    __syncthreads();
  }
  for (int i = t; i < 2304; i += 64) P[(long)b * 2304 + i] = expf(la[i]);
}

__global__ void hk_ascore(const float* __restrict__ qs, const float* __restrict__ pc,
                          float* __restrict__ out) {
  int b = blockIdx.x, t = threadIdx.x;
  float s = 0.f;
  for (int i = t; i < 12288; i += 256) {
    float d = qs[(long)b * 12288 + i] - pc[(long)b * 12288 + i];
    s += fmaxf(d, 0.f);
  }
  __shared__ float red[256];
  red[t] = s; __syncthreads();
  for (int o = 128; o > 0; o >>= 1) { if (t < o) red[t] += red[t + o]; __syncthreads(); }
  if (t == 0) out[b] = -red[0];
}

__global__ void hk_sscore(const float* __restrict__ qs, const float* __restrict__ pc,
                          const float* __restrict__ asc, float* __restrict__ out) {
  int b = blockIdx.x, t = threadIdx.x;
  float s = 0.f;
  for (int i = t; i < 12288; i += 256) {
    float d = qs[(long)b * 12288 + i] - pc[(long)b * 12288 + i];
    s += d * d;
  }
  __shared__ float red[256];
  red[t] = s; __syncthreads();
  for (int o = 128; o > 0; o >>= 1) { if (t < o) red[t] += red[t + o]; __syncthreads(); }
  if (t == 0) out[b] = asc[b] - red[0];
}

// Jonker-Volgenant LSA, one thread per batch; cost = -(i<QS ? plan : 0)
__global__ void hk_hungarian(const float* __restrict__ plan, int* __restrict__ cols) {
  int b = blockIdx.x * blockDim.x + threadIdx.x;
  if (b >= BB) return;
  const float* Pb = plan + (long)b * 2304;
  const double INF = 1e18;
  double u[49], v[49], minv[49];
  int p[49], way[49];
  bool used[49];
  for (int j = 0; j <= 48; ++j) { u[j] = 0; v[j] = 0; p[j] = 0; way[j] = 0; }
  for (int i = 1; i <= 48; ++i) {
    p[0] = i; int j0 = 0;
    for (int j = 0; j <= 48; ++j) { minv[j] = INF; used[j] = false; }
    do {
      used[j0] = true; int i0 = p[j0]; double delta = INF; int j1 = -1;
      for (int j = 1; j <= 48; ++j) if (!used[j]) {
        double cost = (i0 - 1 < QSZ) ? -(double)Pb[(i0 - 1) * 48 + (j - 1)] : 0.0;
        double cur = cost - u[i0] - v[j];
        if (cur < minv[j]) { minv[j] = cur; way[j] = j0; }
        if (minv[j] < delta) { delta = minv[j]; j1 = j; }
      }
      for (int j = 0; j <= 48; ++j) {
        if (used[j]) { u[p[j]] += delta; v[j] -= delta; }
        else minv[j] -= delta;
      }
      j0 = j1;
    } while (p[j0] != 0);
    do { int j1 = way[j0]; p[j0] = p[j1]; j0 = j1; } while (j0);
  }
  for (int j = 1; j <= 48; ++j) cols[b * 48 + p[j] - 1] = j - 1;
}

__global__ void hk_ind_nodes(const int* __restrict__ cols, int* __restrict__ renum,
                             int* __restrict__ incl, int total) {
  int i = blockIdx.x * blockDim.x + threadIdx.x;
  if (i >= total) return;
  int g = i / 48, j = i % 48;
  int half = g & 1, bb = g >> 1;
  int size = half ? CSZ : QSZ;
  int hv = half ? cols[bb * 48 + j] : j;
  if (hv < size) {
    int orig = bb * (QSZ + CSZ) + (half ? QSZ : 0) + hv;
    renum[orig] = g * QSZ + j;
    if (j < QSZ) incl[orig] = 1;
  }
}

__global__ void hk_ind_edges(const int* __restrict__ fi, const int* __restrict__ ti,
                             const int* __restrict__ renum, const int* __restrict__ incl,
                             int* __restrict__ ifr, int* __restrict__ ito,
                             float* __restrict__ em, int total) {
  int i = blockIdx.x * blockDim.x + threadIdx.x;
  if (i >= total) return;
  int a = fi[i], b = ti[i];
  int inc = incl[a] & incl[b];
  ifr[i] = inc ? renum[a] : 0;
  ito[i] = inc ? renum[b] : 0;
  em[i] = inc ? 1.f : 0.f;
}

// h0 for symm: ones(.,32) @ Wn + bn  (every row identical)
__global__ void hk_symm_h0(const float* __restrict__ Wn, const float* __restrict__ bn,
                           float* __restrict__ h, int total) {
  int i = blockIdx.x * blockDim.x + threadIdx.x;
  if (i >= total) return;
  int j = i % 256;
  float s = bn[j];
  for (int k = 0; k < 32; ++k) s += Wn[k * 256 + j];
  h[i] = s;
}

// ======================================================== host side
struct PSet {   // fp32 device pointers, dict-insertion order
  const float *Wn, *bn, *We, *be, *mW1, *mb1, *mW2, *mb2,
              *rW1, *rb1, *rW2, *rb2, *uW1, *ub1, *uW2, *ub2,
              *f1W, *f1b, *f2W, *f2b;
};
struct DevW {   // transposed/padded bf16 weights
  bf16 *WnT, *WeT, *mW1T, *mW2T, *rW1T, *rW2T, *uW1T, *uW2T, *f1T, *f2T;
};

static const int kPsz[20] = {8192, 256, 1024, 64, 221184, 384, 147456, 384,
                             221184, 384, 147456, 384, 245760, 384, 98304, 256,
                             65536, 256, 65536, 256};
// insertion-index -> position in jax sorted-key flatten
static const int kSortedPos[20] = {1, 3, 0, 2, 8, 9, 10, 11, 12, 13, 14, 15,
                                   16, 17, 18, 19, 4, 5, 6, 7};

static void fill_pset(PSet& P, void* const* din, const int* sizes, int n_in,
                      int base, int blob_idx) {
  const float* t[20];
  if (n_in >= 44) {
    bool insertion = (sizes[4] == 8192);
    for (int i = 0; i < 20; ++i)
      t[i] = (const float*)din[base + (insertion ? i : kSortedPos[i])];
  } else {                                   // flat blob per params dict
    const float* p = (const float*)din[blob_idx];
    long off = 0;
    for (int i = 0; i < 20; ++i) { t[i] = p + off; off += kPsz[i]; }
  }
  P.Wn = t[0];  P.bn = t[1];  P.We = t[2];  P.be = t[3];
  P.mW1 = t[4]; P.mb1 = t[5]; P.mW2 = t[6]; P.mb2 = t[7];
  P.rW1 = t[8]; P.rb1 = t[9]; P.rW2 = t[10]; P.rb2 = t[11];
  P.uW1 = t[12]; P.ub1 = t[13]; P.uW2 = t[14]; P.ub2 = t[15];
  P.f1W = t[16]; P.f1b = t[17]; P.f2W = t[18]; P.f2b = t[19];
}

extern "C" void kernel_launch(void* const* d_in, const int* in_sizes, int n_in,
                              void* d_out, int out_size, void* d_ws, size_t ws_size,
                              hipStream_t stream) {
  (void)ws_size; (void)out_size;
  const float* nf = (const float*)d_in[0];
  const float* ef = (const float*)d_in[1];
  const int* fi = (const int*)d_in[2];
  const int* ti = (const int*)d_in[3];
  PSet PA, PS;
  fill_pset(PA, d_in, in_sizes, n_in, 4, 4);
  fill_pset(PS, d_in, in_sizes, n_in, 24, 5);

  // ---- workspace carving (256B aligned)
  size_t off = 0;
  auto carve = [&](size_t bytes) -> void* {
    void* p = (char*)d_ws + off;
    off += (bytes + 255) & ~(size_t)255;
    return p;
  };
  auto carveW = [&](DevW& W) {
    W.WnT = (bf16*)carve(256 * 32 * 2);   W.WeT = (bf16*)carve(64 * 32 * 2);
    W.mW1T = (bf16*)carve(384 * 576 * 2); W.mW2T = (bf16*)carve(384 * 384 * 2);
    W.rW1T = (bf16*)carve(384 * 576 * 2); W.rW2T = (bf16*)carve(384 * 384 * 2);
    W.uW1T = (bf16*)carve(384 * 640 * 2); W.uW2T = (bf16*)carve(256 * 384 * 2);
    W.f1T = (bf16*)carve(256 * 256 * 2);  W.f2T = (bf16*)carve(256 * 256 * 2);
  };
  DevW WA, WS; carveW(WA); carveW(WS);
  bf16* Xn = (bf16*)carve((size_t)NNODE * 32 * 2);
  bf16* Xe = (bf16*)carve((size_t)EDG * 32 * 2);
  float* h = (float*)carve((size_t)NNODE * 256 * 4);
  bf16* ebf = (bf16*)carve((size_t)EDG * 64 * 2);
  float* agg = (float*)carve((size_t)NNODE * 384 * 4);
  bf16* Cbuf = (bf16*)carve((size_t)ECHUNK * 576 * 2);   // also reused as U
  bf16* H1 = (bf16*)carve((size_t)ECHUNK * 384 * 2);     // also reused as HU
  float* qsf = (float*)carve((size_t)BB * 48 * 256 * 4);
  float* csf = (float*)carve((size_t)BB * 48 * 256 * 4);
  bf16* qsb = (bf16*)carve((size_t)BB * 48 * 256 * 2);
  bf16* csb = (bf16*)carve((size_t)BB * 48 * 256 * 2);
  bf16* t1b = (bf16*)carve((size_t)BB * 48 * 256 * 2);
  float* tmpf = (float*)carve((size_t)BB * 48 * 256 * 4);
  bf16* tqb = (bf16*)carve((size_t)BB * 48 * 256 * 2);
  bf16* tcb = (bf16*)carve((size_t)BB * 48 * 256 * 2);
  float* plin = (float*)carve((size_t)BB * 2304 * 4);
  float* plan = (float*)carve((size_t)BB * 2304 * 4);
  bf16* planb = (bf16*)carve((size_t)BB * 48 * 64 * 2);
  bf16* cstb = (bf16*)carve((size_t)BB * 256 * 64 * 2);
  float* pc = (float*)carve((size_t)BB * 48 * 256 * 4);
  float* asc = (float*)carve(BB * 4);
  int* cols = (int*)carve(BB * 48 * 4);
  int* renum = (int*)carve(NNODE * 4);
  int* incl = (int*)carve(NNODE * 4);
  int* ifr = (int*)carve(EDG * 4);
  int* ito = (int*)carve(EDG * 4);
  float* emk = (float*)carve(EDG * 4);
  bf16* Ubf = Cbuf;   // aliases: update-MLP stage runs after edge chunks
  bf16* HU = H1;

  // ---- helpers
  auto EL = [&](auto kern, int total, auto... args) {
    kern<<<dim3((total + 255) / 256), dim3(256), 0, stream>>>(args..., total);
  };
  auto gemm = [&](const bf16* A, const bf16* Bt, const float* bias, int M, int N,
                  int K, void* Y, int mode, int relu, const int* si,
                  const float* sm, float* res) {
    dim3 g(M / 16, (N + 16 * GWAVES - 1) / (16 * GWAVES));
    hk_gemm<<<g, dim3(32 * GWAVES), 0, stream>>>(A, Bt, bias, M, N, K, Y, mode,
                                                 relu, si, sm, res);
  };
  auto gemm_nt = [&](const bf16* A, const bf16* Bm, float* out, int M, int N,
                     int K, long sA, long sB, long sO, int batch) {
    dim3 g(M / 16, (N + 16 * GWAVES - 1) / (16 * GWAVES), batch);
    hk_gemm_nt<<<g, dim3(32 * GWAVES), 0, stream>>>(A, Bm, out, M, N, K, sA, sB, sO);
  };
  auto prepW = [&](const float* W, bf16* Wt, int K, int N, int Kpad) {
    EL(hk_castWT, N * Kpad, W, Wt, K, N, Kpad);
  };
  auto prep_set = [&](const PSet& P, const DevW& W) {
    prepW(P.Wn, W.WnT, 32, 256, 32);    prepW(P.We, W.WeT, 16, 64, 32);
    prepW(P.mW1, W.mW1T, 576, 384, 576); prepW(P.mW2, W.mW2T, 384, 384, 384);
    prepW(P.rW1, W.rW1T, 576, 384, 576); prepW(P.rW2, W.rW2T, 384, 384, 384);
    prepW(P.uW1, W.uW1T, 640, 384, 640); prepW(P.uW2, W.uW2T, 384, 256, 384);
    prepW(P.f1W, W.f1T, 256, 256, 256);  prepW(P.f2W, W.f2T, 256, 256, 256);
  };
  prep_set(PA, WA);
  prep_set(PS, WS);

  // one full message-passing step
  auto run_prop = [&](int Nn, const int* fiP, const int* tiP, const float* em,
                      const DevW& W, const PSet& P) {
    hipMemsetAsync(agg, 0, (size_t)Nn * 384 * 4, stream);
    for (int dir = 0; dir < 2; ++dir) {
      const bf16* W1 = dir ? W.rW1T : W.mW1T;
      const bf16* W2 = dir ? W.rW2T : W.mW2T;
      const float* b1 = dir ? P.rb1 : P.mb1;
      const float* b2 = dir ? P.rb2 : P.mb2;
      const int* sc = dir ? fiP : tiP;       // segment_sum(m,ti)+segment_sum(rm,fi)
      for (int c = 0; c < NCHUNK; ++c) {
        int e0 = c * ECHUNK;
        EL(hk_concat_e, ECHUNK * 576, h, ebf, fiP, tiP, Cbuf, e0, dir);
        gemm(Cbuf, W1, b1, ECHUNK, 384, 576, H1, 1, 1, nullptr, nullptr, nullptr);
        gemm(H1, W2, b2, ECHUNK, 384, 384, agg, 2, 0, sc + e0,
             em ? em + e0 : nullptr, nullptr);
      }
    }
    EL(hk_concat_n, Nn * 640, h, agg, Ubf);
    gemm(Ubf, W.uW1T, P.ub1, Nn, 384, 640, HU, 1, 1, nullptr, nullptr, nullptr);
    gemm(HU, W.uW2T, P.ub2, Nn, 256, 384, nullptr, 3, 0, nullptr, nullptr, h);
  };

  // plan pipeline: h(B*rpb,256) -> plan(B,48,48) + pc(B,48,256), fills qsf/csf
  auto run_plan = [&](int rpb, int qvalid, int cvalid, const DevW& W, const PSet& P) {
    EL(hk_buildqc, BB * 48 * 256, h, qsf, qsb, rpb, 0, qvalid);
    EL(hk_buildqc, BB * 48 * 256, h, csf, csb, rpb, QSZ, cvalid);
    // _lrl(q) * qmask
    gemm(qsb, W.f1T, P.f1b, BB * 48, 256, 256, t1b, 1, 1, nullptr, nullptr, nullptr);
    gemm(t1b, W.f2T, P.f2b, BB * 48, 256, 256, tmpf, 0, 0, nullptr, nullptr, nullptr);
    EL(hk_maskcast, BB * 48 * 256, tmpf, tqb, QSZ);
    // _lrl(c) * cmask
    gemm(csb, W.f1T, P.f1b, BB * 48, 256, 256, t1b, 1, 1, nullptr, nullptr, nullptr);
    gemm(t1b, W.f2T, P.f2b, BB * 48, 256, 256, tmpf, 0, 0, nullptr, nullptr, nullptr);
    EL(hk_maskcast, BB * 48 * 256, tmpf, tcb, cvalid);
    // einsum bqd,bcd->bqc ; sinkhorn ; plan @ cs
    gemm_nt(tqb, tcb, plin, 48, 48, 256, 12288, 12288, 2304, BB);
    hk_sinkhorn<<<dim3(BB), dim3(64), 0, stream>>>(plin, plan);
    EL(hk_planpad, BB * 48 * 64, plan, planb);
    EL(hk_cst, BB * 256 * 64, csf, cstb);
    gemm_nt(planb, cstb, pc, 48, 256, 64, 3072, 16384, 12288, BB);
  };

  // ================= ASYMM graph =================
  EL(hk_castpad, NNODE * 32, nf, Xn, DNIN, 32);
  EL(hk_castpad, EDG * 32, ef, Xe, DEIN, 32);
  gemm(Xn, WA.WnT, PA.bn, NNODE, 256, 32, h, 0, 0, nullptr, nullptr, nullptr);
  gemm(Xe, WA.WeT, PA.be, EDG, 64, 32, ebf, 1, 0, nullptr, nullptr, nullptr);
  for (int p = 0; p < NPROP; ++p) run_prop(NNODE, fi, ti, nullptr, WA, PA);
  run_plan(QSZ + CSZ, QSZ, CSZ, WA, PA);
  hk_ascore<<<dim3(BB), dim3(256), 0, stream>>>(qsf, pc, asc);
  hk_hungarian<<<dim3(1), dim3(64), 0, stream>>>(plan, cols);

  // ================= induced structure =================
  hipMemsetAsync(renum, 0, NNODE * 4, stream);
  hipMemsetAsync(incl, 0, NNODE * 4, stream);
  EL(hk_ind_nodes, 2 * BB * 48, cols, renum, incl);
  EL(hk_ind_edges, EDG, fi, ti, renum, incl, ifr, ito, emk);

  // ================= SYMM graph =================
  EL(hk_symm_h0, NIND * 256, PS.Wn, PS.bn, h);
  gemm(Xe, WS.WeT, PS.be, EDG, 64, 32, ebf, 1, 0, nullptr, nullptr, nullptr);
  for (int p = 0; p < NPROP; ++p) run_prop(NIND, ifr, ito, emk, WS, PS);
  run_plan(2 * QSZ, QSZ, QSZ, WS, PS);
  hk_sscore<<<dim3(BB), dim3(256), 0, stream>>>(qsf, pc, asc, (float*)d_out);
}
